// ConvTransE_19696720020158
// MI455X (gfx1250) — matrix operand order repaired and checked by reading, not compiled, not run on hardware
//
#include <hip/hip_runtime.h>
#include <hip/hip_bf16.h>

// ---------- types ----------
typedef __bf16 bf16_t;
typedef bf16_t v16bf __attribute__((ext_vector_type(16)));
typedef bf16_t v4bf  __attribute__((ext_vector_type(4)));
typedef float  v8f   __attribute__((ext_vector_type(8)));
typedef unsigned int v4u __attribute__((ext_vector_type(4)));

union FragU { v16bf v; v4u u[2]; };

// GCC-style vector type matching the async-copy builtin prototype
typedef int g4i __attribute__((vector_size(16)));
typedef __attribute__((address_space(1))) g4i g4i_as1;
typedef __attribute__((address_space(3))) g4i g4i_as3;

// ---------- problem constants ----------
#define NUM_ENT 50000
#define NUM_REL 500
#define DIM     512
#define CCH     50
#define KW      3
#define BATCH   1024
#define FCIN    (DIM * CCH)          // 25600
#define FILM    350                  // (2K+1)*C
#define FILMPAD 704                  // 700 padded to /16
#define SPLIT   300                  // 2*C*K
#define BN1N    (BATCH * DIM)        // 524288 per channel
#define EPSV    1e-5f

// ---------- GEMM tiling ----------
#define BM 64
#define BN 256
#define KSTEP 32
#define LDSTR 40                     // padded row stride (elements) -> conflict-free

#if defined(__gfx1250__) && __has_builtin(__builtin_amdgcn_global_load_async_to_lds_b128) && __has_builtin(__builtin_amdgcn_s_wait_asynccnt)
#define USE_ASYNC 1
#else
#define USE_ASYNC 0
#endif

__device__ __forceinline__ float wave_sum(float v) {
#pragma unroll
  for (int o = 16; o > 0; o >>= 1) v += __shfl_xor(v, o, 32);
  return v;
}

#if USE_ASYNC
__device__ __forceinline__ void async_cp16(const void* g, void* l) {
  __builtin_amdgcn_global_load_async_to_lds_b128(
      (g4i_as1*)(unsigned long long)g,
      (g4i_as3*)(unsigned)(unsigned long long)l,
      0, 0);
}
#endif

// ---------------------------------------------------------------
// 1) e1_all bf16 table: row-wise L2 norm, tanh, convert
// ---------------------------------------------------------------
__global__ __launch_bounds__(256)
void k_tanh_norm_bf16(const float* __restrict__ emb, bf16_t* __restrict__ out) {
  const int wave = threadIdx.x >> 5, lane = threadIdx.x & 31;
  const int row = blockIdx.x * 8 + wave;
  if (row >= NUM_ENT) return;
  const float4* r4 = (const float4*)(emb + (size_t)row * DIM);
  float4 v[4]; float ss = 0.f;
#pragma unroll
  for (int i = 0; i < 4; ++i) {
    v[i] = r4[lane + 32 * i];
    ss += v[i].x * v[i].x + v[i].y * v[i].y + v[i].z * v[i].z + v[i].w * v[i].w;
  }
  ss = wave_sum(ss);
  const float inv = 1.0f / fmaxf(sqrtf(ss), 1e-12f);
  bf16_t* op = out + (size_t)row * DIM;
#pragma unroll
  for (int i = 0; i < 4; ++i) {
    v4bf o;
    o[0] = (bf16_t)tanhf(v[i].x * inv);
    o[1] = (bf16_t)tanhf(v[i].y * inv);
    o[2] = (bf16_t)tanhf(v[i].z * inv);
    o[3] = (bf16_t)tanhf(v[i].w * inv);
    *(v4bf*)(op + (lane + 32 * i) * 4) = o;
  }
}

// ---------------------------------------------------------------
// 2) gather: e1 (recomputed f32), rel (f32), cat (bf16) per triplet
// ---------------------------------------------------------------
__global__ __launch_bounds__(256)
void k_gather(const float* __restrict__ emb, const float* __restrict__ emb_rel,
              const int* __restrict__ trip,
              float* __restrict__ e1, float* __restrict__ rel,
              bf16_t* __restrict__ cat) {
  const int wave = threadIdx.x >> 5, lane = threadIdx.x & 31;
  const int b = blockIdx.x * 8 + wave;
  if (b >= BATCH) return;
  const int t0 = trip[b * 3 + 0];
  const int t1 = trip[b * 3 + 1];
  const float4* r4 = (const float4*)(emb + (size_t)t0 * DIM);
  float4 v[4]; float ss = 0.f;
#pragma unroll
  for (int i = 0; i < 4; ++i) {
    v[i] = r4[lane + 32 * i];
    ss += v[i].x * v[i].x + v[i].y * v[i].y + v[i].z * v[i].z + v[i].w * v[i].w;
  }
  ss = wave_sum(ss);
  const float inv = 1.0f / fmaxf(sqrtf(ss), 1e-12f);
#pragma unroll
  for (int i = 0; i < 4; ++i) {
    float4 t;
    t.x = tanhf(v[i].x * inv); t.y = tanhf(v[i].y * inv);
    t.z = tanhf(v[i].z * inv); t.w = tanhf(v[i].w * inv);
    ((float4*)(e1 + (size_t)b * DIM))[lane + 32 * i] = t;
    v4bf o; o[0]=(bf16_t)t.x; o[1]=(bf16_t)t.y; o[2]=(bf16_t)t.z; o[3]=(bf16_t)t.w;
    *(v4bf*)(cat + (size_t)b * 2 * DIM + (lane + 32 * i) * 4) = o;
  }
  const float4* q4 = (const float4*)(emb_rel + (size_t)t1 * DIM);
#pragma unroll
  for (int i = 0; i < 4; ++i) {
    float4 t = q4[lane + 32 * i];
    ((float4*)(rel + (size_t)b * DIM))[lane + 32 * i] = t;
    v4bf o; o[0]=(bf16_t)t.x; o[1]=(bf16_t)t.y; o[2]=(bf16_t)t.z; o[3]=(bf16_t)t.w;
    *(v4bf*)(cat + (size_t)b * 2 * DIM + DIM + (lane + 32 * i) * 4) = o;
  }
}

// ---------------------------------------------------------------
// 3) BN over stacked (2 channels)
// ---------------------------------------------------------------
__global__ __launch_bounds__(256)
void k_bn1_partial(const float* __restrict__ e1, const float* __restrict__ rel,
                   float2* __restrict__ part) {
  const int ch = blockIdx.y;
  const float* p = ch ? rel : e1;
  float s = 0.f, s2 = 0.f;
  for (int i = blockIdx.x * 256 + threadIdx.x; i < BN1N; i += 32 * 256) {
    float v = p[i]; s += v; s2 += v * v;
  }
  __shared__ float ls[8], ls2[8];
  const int wave = threadIdx.x >> 5, lane = threadIdx.x & 31;
  s = wave_sum(s); s2 = wave_sum(s2);
  if (lane == 0) { ls[wave] = s; ls2[wave] = s2; }
  __syncthreads();
  if (threadIdx.x == 0) {
    float a = 0.f, a2 = 0.f;
#pragma unroll
    for (int w = 0; w < 8; ++w) { a += ls[w]; a2 += ls2[w]; }
    part[ch * 32 + blockIdx.x] = make_float2(a, a2);
  }
}

__global__ __launch_bounds__(64)
void k_bn1_final(const float2* __restrict__ part, float* __restrict__ stats) {
  const int ch = threadIdx.x >> 5, lane = threadIdx.x & 31;
  float2 p = part[ch * 32 + lane];
  float s = wave_sum(p.x), s2 = wave_sum(p.y);
  if (lane == 0) {
    const float mean = s / (float)BN1N;
    const float var = s2 / (float)BN1N - mean * mean;
    stats[ch * 2 + 0] = mean;
    stats[ch * 2 + 1] = rsqrtf(var + EPSV);
  }
}

// ---------------------------------------------------------------
// 4) FiLM weights -> N-major bf16 [704 x 1024]
// ---------------------------------------------------------------
__global__ __launch_bounds__(256)
void k_prep_wT(const float* __restrict__ sw, const float* __restrict__ shw,
               bf16_t* __restrict__ wT) {
  int t = blockIdx.x * 256 + threadIdx.x;
  if (t >= FILMPAD * 2 * DIM) return;
  int n = t >> 10, k = t & 1023;
  float v = 0.f;
  if (n < FILM)            v = sw[(size_t)k * FILM + n];
  else if (n < 2 * FILM)   v = shw[(size_t)k * FILM + (n - FILM)];
  wT[t] = (bf16_t)v;
}

// ---------------------------------------------------------------
// Generic NT bf16 GEMM, f32 accumulate, LDS double-buffered async staging.
//   C[M,N] = A[M,K] * B[N,K]^T (+ bias[n])
//   block = 256 thr (8 waves); block tile 64M x 256N; wave tile 64M x 32N.
//   Per stage (K-step 32): 1280 16B chunks staged (A:256, B:1024) = 5/thread.
// ---------------------------------------------------------------
__global__ __launch_bounds__(256)
void k_gemm_bf16_nt(const bf16_t* __restrict__ A, const bf16_t* __restrict__ B,
                    float* __restrict__ Cmat, const float* __restrict__ bias,
                    int N, int K, int lda, int ldb, int ldc) {
  __shared__ __align__(16) bf16_t lsA[2][BM * LDSTR];
  __shared__ __align__(16) bf16_t lsB[2][BN * LDSTR];

  const int tid  = threadIdx.x;
  const int wave = tid >> 5, lane = tid & 31;
  const int rsel = lane & 15, hi = lane >> 4;
  const int m0 = blockIdx.y * BM;
  const int nb = blockIdx.x * BN;
  const int n0 = nb + wave * 32;

  // staging assignment (k-invariant parts)
  const int arow = tid >> 2;                 // A: 64 rows x 4 chunks
  const int acol = (tid & 3) * 8;
  const bf16_t* agp = A + (size_t)(m0 + arow) * lda + acol;
  const int aoff = arow * LDSTR + acol;
  int brow[4], boff[4];
  const bf16_t* bgp[4];
#pragma unroll
  for (int j = 0; j < 4; ++j) {
    const int c2 = tid + 256 * j;            // B: 256 rows x 4 chunks
    brow[j] = c2 >> 2;
    const int bcol = (c2 & 3) * 8;
    const int grow = min(nb + brow[j], N - 1);   // clamp OOB rows (results unused)
    bgp[j] = B + (size_t)grow * ldb + bcol;
    boff[j] = brow[j] * LDSTR + bcol;
  }

  const int stages = K / KSTEP;
  v8f acc[4][2] = {};

#if USE_ASYNC
  // issue stage 0
  {
    async_cp16(agp, &lsA[0][aoff]);
#pragma unroll
    for (int j = 0; j < 4; ++j) async_cp16(bgp[j], &lsB[0][boff[j]]);
  }
#endif

  for (int s = 0; s < stages; ++s) {
    const int buf = s & 1;
#if USE_ASYNC
    __builtin_amdgcn_s_wait_asynccnt(0);     // this wave's stage-s chunks in LDS
    __syncthreads();                          // whole tile visible
    if (s + 1 < stages) {
      const int k1 = (s + 1) * KSTEP;
      async_cp16(agp + k1, &lsA[buf ^ 1][aoff]);
#pragma unroll
      for (int j = 0; j < 4; ++j) async_cp16(bgp[j] + k1, &lsB[buf ^ 1][boff[j]]);
    }
#else
    const int k0 = s * KSTEP;
    v4u ra = *(const v4u*)(agp + k0);
    v4u rb[4];
#pragma unroll
    for (int j = 0; j < 4; ++j) rb[j] = *(const v4u*)(bgp[j] + k0);
    __syncthreads();                          // previous stage fully consumed
    *(v4u*)(&lsA[0][aoff]) = ra;
#pragma unroll
    for (int j = 0; j < 4; ++j) *(v4u*)(&lsB[0][boff[j]]) = rb[j];
    __syncthreads();
#endif

    // compute stage s
    const bf16_t* bufA = &lsA[USE_ASYNC ? buf : 0][0];
    const bf16_t* bufB = &lsB[USE_ASYNC ? buf : 0][0];
    FragU af[4];
#pragma unroll
    for (int q = 0; q < 4; ++q) {            // A 16x32 frag layout
      const bf16_t* ap = bufA + (q * 16 + rsel) * LDSTR;
      af[q].u[0] = *(const v4u*)(ap + hi * 8);
      af[q].u[1] = *(const v4u*)(ap + 16 + hi * 8);
    }
#pragma unroll
    for (int t = 0; t < 2; ++t) {            // B 32x16 frag layout
      FragU bfr;
      const bf16_t* bp = bufB + (wave * 32 + t * 16 + rsel) * LDSTR;
      bfr.u[0] = *(const v4u*)(bp + hi * 8);
      bfr.u[1] = *(const v4u*)(bp + 16 + hi * 8);
#pragma unroll
      for (int q = 0; q < 4; ++q)
        acc[q][t] = __builtin_amdgcn_wmma_f32_16x16x32_bf16(
            false, af[q].v, false, bfr.v, (short)0, acc[q][t], false, false);
    }
  }

  // epilogue
#pragma unroll
  for (int t = 0; t < 2; ++t) {
    const int ncol0 = n0 + t * 16;
    if (ncol0 >= N) continue;                // wave-uniform
    const int col = ncol0 + rsel;
    const float bs = bias ? bias[col] : 0.0f;
#pragma unroll
    for (int q = 0; q < 4; ++q) {
      const int row = m0 + q * 16 + hi * 8;
      float* cp = Cmat + (size_t)row * ldc + col;
#pragma unroll
      for (int r = 0; r < 8; ++r) cp[(size_t)r * ldc] = acc[q][t][r] + bs;
    }
  }
}

// ---------------------------------------------------------------
// 5) FiLM apply
// ---------------------------------------------------------------
__global__ __launch_bounds__(352)
void k_film_apply(const float* __restrict__ ab, const float* __restrict__ scale_b,
                  const float* __restrict__ shift_b, const float* __restrict__ th_w,
                  const float* __restrict__ th_b,
                  float* __restrict__ wf, float* __restrict__ bf_) {
  const int b = blockIdx.x, j = threadIdx.x;
  if (j >= FILM) return;
  const float a  = ab[(size_t)b * FILMPAD + j] + scale_b[j];
  const float be = ab[(size_t)b * FILMPAD + FILM + j] + shift_b[j];
  if (j < SPLIT) wf[b * SPLIT + j] = (a + 1.0f) * th_w[j] + be;
  else           bf_[b * CCH + (j - SPLIT)] = (a + 1.0f) * th_b[j - SPLIT] + be;
}

// ---------------------------------------------------------------
// 6) per-sample conv (C=50,K=3) fused with BN1 -> y bf16
// ---------------------------------------------------------------
__global__ __launch_bounds__(256)
void k_conv(const float* __restrict__ e1, const float* __restrict__ rel,
            const float* __restrict__ stats, const float* __restrict__ wf,
            const float* __restrict__ bf_, bf16_t* __restrict__ y) {
  __shared__ float xp[2][DIM + 2];
  __shared__ float wl[SPLIT];
  __shared__ float bl[CCH];
  const int b = blockIdx.x, tid = threadIdx.x;
  const float m0 = stats[0], i0 = stats[1], m1 = stats[2], i1 = stats[3];
  for (int d = tid; d < DIM; d += 256) {
    xp[0][d + 1] = (e1[(size_t)b * DIM + d] - m0) * i0;
    xp[1][d + 1] = (rel[(size_t)b * DIM + d] - m1) * i1;
  }
  if (tid < 2) { xp[tid][0] = 0.f; xp[tid][DIM + 1] = 0.f; }
  for (int j = tid; j < SPLIT; j += 256) wl[j] = wf[b * SPLIT + j];
  if (tid < CCH) bl[tid] = bf_[b * CCH + tid];
  __syncthreads();
  for (int idx = tid; idx < CCH * DIM; idx += 256) {
    const int c = idx >> 9, j = idx & (DIM - 1);
    const float* wc = &wl[c * 6];
    float acc = bl[c];
    acc += xp[0][j] * wc[0] + xp[0][j + 1] * wc[1] + xp[0][j + 2] * wc[2];
    acc += xp[1][j] * wc[3] + xp[1][j + 1] * wc[4] + xp[1][j + 2] * wc[5];
    y[(size_t)b * FCIN + idx] = (bf16_t)acc;
  }
}

// ---------------------------------------------------------------
// 7) fc_w f32 -> bf16
// ---------------------------------------------------------------
__global__ __launch_bounds__(256)
void k_fcw_bf16(const float* __restrict__ w, bf16_t* __restrict__ o) {
  const int t = blockIdx.x * 256 + threadIdx.x;
  if (t >= (DIM * FCIN) / 4) return;
  float4 v = ((const float4*)w)[t];
  v4bf r; r[0]=(bf16_t)v.x; r[1]=(bf16_t)v.y; r[2]=(bf16_t)v.z; r[3]=(bf16_t)v.w;
  *(v4bf*)(o + (size_t)t * 4) = r;
}

// ---------------------------------------------------------------
// 8) BN over batch + ReLU -> h bf16
// ---------------------------------------------------------------
__global__ __launch_bounds__(256)
void k_bn2_stats(const float* __restrict__ h, float* __restrict__ st) {
  const int n = blockIdx.x * 256 + threadIdx.x;
  float s = 0.f, s2 = 0.f;
  for (int r = 0; r < BATCH; ++r) {
    float v = h[(size_t)r * DIM + n]; s += v; s2 += v * v;
  }
  const float mean = s / (float)BATCH;
  const float var = s2 / (float)BATCH - mean * mean;
  st[n] = mean;
  st[DIM + n] = rsqrtf(var + EPSV);
}

__global__ __launch_bounds__(256)
void k_bn2_apply(const float* __restrict__ h, const float* __restrict__ st,
                 bf16_t* __restrict__ hb) {
  const int e = blockIdx.x * 256 + threadIdx.x;
  if (e >= BATCH * DIM) return;
  const int n = e & (DIM - 1);
  const float v = (h[e] - st[n]) * st[DIM + n];
  hb[e] = (bf16_t)fmaxf(v, 0.0f);
}

// ---------------------------------------------------------------
// launch
// ---------------------------------------------------------------
extern "C" void kernel_launch(void* const* d_in, const int* in_sizes, int n_in,
                              void* d_out, int out_size, void* d_ws, size_t ws_size,
                              hipStream_t stream) {
  (void)in_sizes; (void)n_in; (void)out_size; (void)ws_size;
  const float* embedding = (const float*)d_in[0];
  const float* emb_rel   = (const float*)d_in[1];
  const float* scale_w   = (const float*)d_in[3];
  const float* scale_b   = (const float*)d_in[4];
  const float* shift_w   = (const float*)d_in[5];
  const float* shift_b   = (const float*)d_in[6];
  const float* theta_w   = (const float*)d_in[7];
  const float* theta_b   = (const float*)d_in[8];
  const float* fc_w      = (const float*)d_in[9];
  const float* fc_b      = (const float*)d_in[10];
  const int*   triplets  = (const int*)d_in[11];
  float* out = (float*)d_out;

  char* p = (char*)d_ws;
  auto carve = [&](size_t bytes) -> void* {
    void* r = (void*)p; p += (bytes + 255) & ~(size_t)255; return r;
  };
  bf16_t* e1_all_bf = (bf16_t*)carve((size_t)NUM_ENT * DIM * 2);
  bf16_t* y_bf      = (bf16_t*)carve((size_t)BATCH * FCIN * 2);
  bf16_t* fcw_bf    = (bf16_t*)carve((size_t)DIM * FCIN * 2);
  bf16_t* cat_bf    = (bf16_t*)carve((size_t)BATCH * 2 * DIM * 2);
  bf16_t* wT_bf     = (bf16_t*)carve((size_t)FILMPAD * 2 * DIM * 2);
  bf16_t* h_bf      = (bf16_t*)carve((size_t)BATCH * DIM * 2);
  float*  e1        = (float*)carve((size_t)BATCH * DIM * 4);
  float*  rel       = (float*)carve((size_t)BATCH * DIM * 4);
  float*  ab        = (float*)carve((size_t)BATCH * FILMPAD * 4);
  float*  wf        = (float*)carve((size_t)BATCH * SPLIT * 4);
  float*  bfilm     = (float*)carve((size_t)BATCH * CCH * 4);
  float*  h_pre     = (float*)carve((size_t)BATCH * DIM * 4);
  float2* bn1part   = (float2*)carve(2 * 32 * sizeof(float2));
  float*  bn1stats  = (float*)carve(4 * 4);
  float*  bn2stats  = (float*)carve(2 * DIM * 4);

  k_tanh_norm_bf16<<<NUM_ENT / 8, 256, 0, stream>>>(embedding, e1_all_bf);
  k_gather<<<BATCH / 8, 256, 0, stream>>>(embedding, emb_rel, triplets, e1, rel, cat_bf);
  k_bn1_partial<<<dim3(32, 2), 256, 0, stream>>>(e1, rel, bn1part);
  k_bn1_final<<<1, 64, 0, stream>>>(bn1part, bn1stats);
  k_prep_wT<<<(FILMPAD * 2 * DIM + 255) / 256, 256, 0, stream>>>(scale_w, shift_w, wT_bf);
  // FiLM GEMM: M=1024, N=704, K=1024
  k_gemm_bf16_nt<<<dim3(3, 16), 256, 0, stream>>>(
      cat_bf, wT_bf, ab, (const float*)nullptr,
      FILMPAD, 2 * DIM, 2 * DIM, 2 * DIM, FILMPAD);
  k_film_apply<<<BATCH, 352, 0, stream>>>(ab, scale_b, shift_b, theta_w, theta_b, wf, bfilm);
  k_conv<<<BATCH, 256, 0, stream>>>(e1, rel, bn1stats, wf, bfilm, y_bf);
  k_fcw_bf16<<<(DIM * FCIN / 4 + 255) / 256, 256, 0, stream>>>(fc_w, fcw_bf);
  // fc GEMM: M=1024, N=512, K=25600 (+fc_b)
  k_gemm_bf16_nt<<<dim3(2, 16), 256, 0, stream>>>(
      y_bf, fcw_bf, h_pre, fc_b, DIM, FCIN, FCIN, FCIN, DIM);
  k_bn2_stats<<<2, 256, 0, stream>>>(h_pre, bn2stats);
  k_bn2_apply<<<(BATCH * DIM + 255) / 256, 256, 0, stream>>>(h_pre, bn2stats, h_bf);
  // score GEMM: M=1024, N=50000, K=512
  k_gemm_bf16_nt<<<dim3((NUM_ENT + BN - 1) / BN, 16), 256, 0, stream>>>(
      h_bf, e1_all_bf, out, (const float*)nullptr,
      NUM_ENT, DIM, DIM, DIM, NUM_ENT);
}
